// DilatedWindowMSA_25443386261906
// MI455X (gfx1250) — compile-verified, matrix operand order checked by reading, not run on hardware
//
#include <hip/hip_runtime.h>
#include <hip/hip_bf16.h>

typedef __attribute__((ext_vector_type(16))) _Float16 v16h;
typedef __attribute__((ext_vector_type(2)))  __fp16   v2fp16;   // cvt_pkrtz result type
typedef __attribute__((ext_vector_type(8)))  float    v8f;

#define N_TOK 49
#define EMB   256
#define HEADS 8
#define HD    32
#define SCALE 0.17677669529663687f   // 32^-0.5

// ---------------- workspace layout (bytes) ----------------
#define WS_WQKV   0           // 768*256 f16 = 393216
#define WS_WPROJ  393216      // 256*256 f16 = 131072
#define WS_QKVB   524288      // 768 f32     = 3072
#define WS_BIAS   528384      // 8*64*64 f32 = 131072

// ---------------- prep kernel: f16 weights + padded bias ----------------
__global__ void swin_prep_kernel(const float* __restrict__ qkv_w,
                                 const float* __restrict__ qkv_b,
                                 const float* __restrict__ proj_w,
                                 const float* __restrict__ bias_table,
                                 const int*   __restrict__ rel_index,
                                 _Float16* __restrict__ wqkv,
                                 _Float16* __restrict__ wproj,
                                 float*    __restrict__ qkvb,
                                 float*    __restrict__ biasPad) {
    const int tid    = blockIdx.x * blockDim.x + threadIdx.x;
    const int stride = gridDim.x * blockDim.x;

    for (int i = tid; i < 768 * 256; i += stride) {
        const int row = i >> 8;                       // output column of QKV
        const float s = (row < 256) ? SCALE : 1.0f;   // fold q-scale into weights
        wqkv[i] = (_Float16)(qkv_w[i] * s);
    }
    for (int i = tid; i < 256 * 256; i += stride)
        wproj[i] = (_Float16)proj_w[i];
    for (int i = tid; i < 768; i += stride)
        qkvb[i] = qkv_b[i] * ((i < 256) ? SCALE : 1.0f);
    for (int i = tid; i < HEADS * 64 * 64; i += stride) {
        const int h = i >> 12, n = (i >> 6) & 63, m = i & 63;
        float v;
        if (m >= N_TOK)      v = -1.0e30f;            // mask padded keys
        else if (n >= N_TOK) v = 0.0f;                // padded query rows: don't care
        else                 v = bias_table[rel_index[n * N_TOK + m] * HEADS + h];
        biasPad[i] = v;
    }
}

// ---------------- helpers ----------------
__device__ __forceinline__ v16h ldfrag(const _Float16* p) {
    union { v16h v; uint4 u[2]; } t;
    t.u[0] = *(const uint4*)p;
    t.u[1] = *(const uint4*)(p + 16);
    return t.v;
}

__device__ __forceinline__ unsigned pk2(float x, float y) {
    union { v2fp16 h; unsigned u; } t;
    t.h = __builtin_amdgcn_cvt_pkrtz(x, y);   // v_cvt_pk_rtz_f16_f32
    return t.u;
}

__device__ __forceinline__ v8f wmma_f16(v16h a, v16h b, v8f c) {
    return __builtin_amdgcn_wmma_f32_16x16x32_f16(false, a, false, b,
                                                  (short)0, c, false, false);
}

// ---------------- main kernel: one workgroup per window ----------------
// LDS (128 KB total -> 2 workgroups per WGP):
//   sX / sXout : 64x256 f16                       (32 KB)  halves [0, 16384)
//   per-head   : q 64x32 | k 64x32  (8 KB/head)   (64 KB)  halves [16384, 49152)
//                -- after softmax, the same 8 KB block is reused as attn 64x64 f16
//   sVT        : 8 x 32x64 f16 (V transposed)     (32 KB)  halves [49152, 65536)
__global__ void __launch_bounds__(256)
swin_msa_kernel(const float* __restrict__ X,
                const _Float16* __restrict__ wqkv,
                const _Float16* __restrict__ wproj,
                const float* __restrict__ qkvb,
                const float* __restrict__ biasPad,
                const float* __restrict__ proj_b,
                float* __restrict__ out) {
    extern __shared__ _Float16 smem[];
    _Float16* sX  = smem;                       // aliased as sXout after barrier

    const int lane  = threadIdx.x & 31;
    const int h     = threadIdx.x >> 5;         // wave id == head id
    const int laneN = lane & 15;
    const int kro   = (lane & 16) ? 8 : 0;      // K-chunk select (A/B) / row offset (C)

    _Float16* sQ  = smem + 16384 + h * 4096;    // 64x32, row stride 32 halves
    _Float16* sK  = sQ + 2048;                  // 64x32
    _Float16* sA  = sQ;                         // 64x64 attn, aliases q+k after softmax
    _Float16* sVT = smem + 49152 + h * 2048;    // 32x64, row stride 64 halves

    // ---- Phase A: stage X into LDS as f16, zero-pad rows 49..63 ----
    const float* Xb = X + (size_t)blockIdx.x * (N_TOK * EMB);
    for (int v = threadIdx.x; v < 64 * 64; v += 256) {       // 64 rows x 64 float4
        const int row = v >> 6, c4 = v & 63;
        float4 val = make_float4(0.f, 0.f, 0.f, 0.f);
        if (row < N_TOK) val = ((const float4*)Xb)[row * 64 + c4];
        uint2 p = make_uint2(pk2(val.x, val.y), pk2(val.z, val.w));
        *(uint2*)(sX + row * 256 + c4 * 4) = p;              // ds_store_b64
    }
    __syncthreads();

    // ---- Phase B: QKV projection for this wave's head ----
    // col tiles: q {2h,2h+1}, k {16+2h,17+2h}, v {32+2h,33+2h}
    for (int t = 0; t < 6; ++t) {
        const int ctg  = ((t >> 1) << 4) + 2 * h + (t & 1);
        const int wrow = ctg * 16 + laneN;                   // weight row == out col
        __builtin_prefetch(wqkv + wrow * 256, 0, 1);         // global_prefetch_b8
        v8f acc[4];
        { v8f z = {}; acc[0] = z; acc[1] = z; acc[2] = z; acc[3] = z; }
        for (int kk = 0; kk < 8; ++kk) {
            const v16h bf = ldfrag(wqkv + wrow * 256 + kk * 32 + kro);
            for (int mt = 0; mt < 4; ++mt) {
                const v16h af = ldfrag(sX + (mt * 16 + laneN) * 256 + kk * 32 + kro);
                acc[mt] = wmma_f16(af, bf, acc[mt]);
            }
        }
        const float bb = qkvb[wrow];
        const int   d  = (t & 1) * 16 + laneN;               // 0..31 col within head
        for (int mt = 0; mt < 4; ++mt) {
            if (ctg < 16) {                                  // q: token-major rows
                for (int r = 0; r < 8; ++r)
                    sQ[(mt * 16 + r + kro) * 32 + d] = (_Float16)(acc[mt][r] + bb);
            } else if (ctg < 32) {                           // k: token-major rows
                for (int r = 0; r < 8; ++r)
                    sK[(mt * 16 + r + kro) * 32 + d] = (_Float16)(acc[mt][r] + bb);
            } else {                                         // v: transposed, M-contig
                const int Mb = mt * 16 + kro;                // 16B aligned
                uint4 p;
                p.x = pk2(acc[mt][0] + bb, acc[mt][1] + bb);
                p.y = pk2(acc[mt][2] + bb, acc[mt][3] + bb);
                p.z = pk2(acc[mt][4] + bb, acc[mt][5] + bb);
                p.w = pk2(acc[mt][6] + bb, acc[mt][7] + bb);
                *(uint4*)(sVT + d * 64 + Mb) = p;            // ds_store_b128
            }
        }
    }
    __syncthreads();   // everyone done reading sX; it becomes sXout below

    // ---- Phase C: all 16 score tiles in registers (bias as WMMA C), softmax ----
    v8f att[16];
    for (int mt = 0; mt < 4; ++mt)
        for (int nt = 0; nt < 4; ++nt)
            for (int r = 0; r < 8; ++r)
                att[mt * 4 + nt][r] =
                    biasPad[h * 4096 + (mt * 16 + r + kro) * 64 + nt * 16 + laneN];
    for (int mt = 0; mt < 4; ++mt) {
        const v16h qf = ldfrag(sQ + (mt * 16 + laneN) * 32 + kro);
        for (int nt = 0; nt < 4; ++nt) {
            const v16h kf = ldfrag(sK + (nt * 16 + laneN) * 32 + kro);
            att[mt * 4 + nt] = wmma_f16(qf, kf, att[mt * 4 + nt]);
        }
    }
    // softmax: row = (mt, r, lane-half) spread over 4 nt-tiles x 16 lanes,
    // then store attn into the (now dead) q/k block of this head.
    for (int mt = 0; mt < 4; ++mt) {
        for (int r = 0; r < 8; ++r) {
            float mx = fmaxf(fmaxf(att[mt * 4 + 0][r], att[mt * 4 + 1][r]),
                             fmaxf(att[mt * 4 + 2][r], att[mt * 4 + 3][r]));
            for (int off = 8; off; off >>= 1) mx = fmaxf(mx, __shfl_xor(mx, off, 16));
            float s = 0.f;
            for (int nt = 0; nt < 4; ++nt) {
                const float e = __expf(att[mt * 4 + nt][r] - mx);
                att[mt * 4 + nt][r] = e; s += e;
            }
            for (int off = 8; off; off >>= 1) s += __shfl_xor(s, off, 16);
            const float inv = 1.0f / s;
            const int n = mt * 16 + r + kro;
            for (int nt = 0; nt < 4; ++nt)
                sA[n * 64 + nt * 16 + laneN] = (_Float16)(att[mt * 4 + nt][r] * inv);
        }
    }

    // ---- Phase C2: attn @ V  -> sXout (aliases sX) ----
    _Float16* sXout = sX;
    for (int mt = 0; mt < 4; ++mt) {
        for (int nd = 0; nd < 2; ++nd) {
            v8f acc = {};
            for (int kk = 0; kk < 2; ++kk) {
                const v16h af = ldfrag(sA + (mt * 16 + laneN) * 64 + kk * 32 + kro);
                const v16h vf = ldfrag(sVT + (nd * 16 + laneN) * 64 + kk * 32 + kro);
                acc = wmma_f16(af, vf, acc);
            }
            const int col = h * 32 + nd * 16 + laneN;
            for (int r = 0; r < 8; ++r)
                sXout[(mt * 16 + r + kro) * 256 + col] = (_Float16)acc[r];
        }
    }
    __syncthreads();   // heads mixed from here

    // ---- Phase D: output projection, f32 store to global ----
    float* Ob = out + (size_t)blockIdx.x * (N_TOK * EMB);
    for (int n2 = 0; n2 < 2; ++n2) {
        const int col = (h * 2 + n2) * 16 + laneN;
        v8f acc[4];
        { v8f z = {}; acc[0] = z; acc[1] = z; acc[2] = z; acc[3] = z; }
        for (int kk = 0; kk < 8; ++kk) {
            const v16h bf = ldfrag(wproj + col * 256 + kk * 32 + kro);
            for (int mt = 0; mt < 4; ++mt) {
                const v16h af = ldfrag(sXout + (mt * 16 + laneN) * 256 + kk * 32 + kro);
                acc[mt] = wmma_f16(af, bf, acc[mt]);
            }
        }
        const float pb = proj_b[col];
        for (int mt = 0; mt < 4; ++mt)
            for (int r = 0; r < 8; ++r) {
                const int M = mt * 16 + r + kro;
                if (M < N_TOK) Ob[M * 256 + col] = acc[mt][r] + pb;
            }
    }
}

extern "C" void kernel_launch(void* const* d_in, const int* in_sizes, int n_in,
                              void* d_out, int out_size, void* d_ws, size_t ws_size,
                              hipStream_t stream) {
    const float* X          = (const float*)d_in[0];
    const float* qkv_w      = (const float*)d_in[1];
    const float* qkv_b      = (const float*)d_in[2];
    const float* proj_w     = (const float*)d_in[3];
    const float* proj_b     = (const float*)d_in[4];
    const float* bias_table = (const float*)d_in[5];
    const int*   rel_index  = (const int*)d_in[6];
    float* out = (float*)d_out;

    char* ws = (char*)d_ws;
    _Float16* wqkv    = (_Float16*)(ws + WS_WQKV);
    _Float16* wproj   = (_Float16*)(ws + WS_WPROJ);
    float*    qkvb    = (float*)(ws + WS_QKVB);
    float*    biasPad = (float*)(ws + WS_BIAS);

    const int B = in_sizes[0] / (N_TOK * EMB);   // 4096 windows

    swin_prep_kernel<<<64, 256, 0, stream>>>(qkv_w, qkv_b, proj_w, bias_table,
                                             rel_index, wqkv, wproj, qkvb, biasPad);

    const size_t ldsBytes = 131072;              // 128 KB -> 2 workgroups / WGP
    swin_msa_kernel<<<B, 256, ldsBytes, stream>>>(X, wqkv, wproj, qkvb, biasPad,
                                                  proj_b, out);
}
